// TabPFNBackbone_43911745635226
// MI455X (gfx1250) — matrix operand order repaired and checked by reading, not compiled, not run on hardware
//
#include <hip/hip_runtime.h>

typedef unsigned short u16;
typedef __attribute__((ext_vector_type(16))) __bf16 v16bf;
typedef __attribute__((ext_vector_type(8)))  __bf16 v8bf;
typedef __attribute__((ext_vector_type(8)))  float  v8f;

#define Bb   4
#define Rr   1024
#define Gg   8
#define Ee   192
#define Ll   4
#define Hh   6
#define Dd   32
#define Ff   384
#define SEP  768
#define NT   (Bb*Rr*Gg)   /* 32768 tokens */
#define BRn  (Bb*Rr)      /* 4096 feature-attn sequences */
#define BGn  (Bb*Gg)      /* 32 row-attn batches */

// ---------- bf16 helpers ----------
__device__ __forceinline__ float bf2f(u16 b) {
  unsigned u = ((unsigned)b) << 16;
  return __builtin_bit_cast(float, u);
}
__device__ __forceinline__ void st_bf16(u16* p, float v) {
  *(__bf16*)p = (__bf16)v;   // native RTNE convert + 16-bit store
}
// 8 contiguous bf16 as one 16B vector load
__device__ __forceinline__ v8bf ld8(const u16* p) {
  return __builtin_bit_cast(v8bf, *(const uint4*)p);
}
__device__ __forceinline__ v16bf mk16(v8bf a, v8bf b) {
  return __builtin_shufflevector(a, b, 0,1,2,3,4,5,6,7,8,9,10,11,12,13,14,15);
}

// ---------- weight convert + transpose: W[L,K,N] f32 -> Wt[L,N,K] bf16 ----------
__global__ void cvt_tr_kernel(const float* __restrict__ in, u16* __restrict__ out,
                              int Kd, int Nd, int total) {
  int i = blockIdx.x * blockDim.x + threadIdx.x;
  if (i >= total) return;
  int kn = Kd * Nd;
  int l = i / kn, rem = i - l * kn;
  int k = rem / Nd, n = rem - k * Nd;
  st_bf16(&out[(size_t)l * kn + (size_t)n * Kd + k], in[i]);
}

// ---------- copy f32 + emit bf16 copy (initial activation) ----------
__global__ void copy_cvt_kernel(const float* __restrict__ in, float* __restrict__ outF,
                                u16* __restrict__ outB, int n) {
  int i = blockIdx.x * blockDim.x + threadIdx.x;
  if (i < n) { float v = in[i]; outF[i] = v; st_bf16(&outB[i], v); }
}

// ---------- LDS-free WMMA bf16 GEMM: C = act(A@W + res) ----------
// A: [M,K] bf16 row-major; Wt: [N,K] bf16 (transposed weights);
// res f32 [M,N] (nullable); outF f32 / outB bf16 (nullable);
// outVT (nullable): store transposed as [bg][h][d][R] (M interpreted as [BG,R], N as [H,D]).
// Each wave: 32x32 tile (4 WMMAs/k-step); block of 4 waves: 64x64 tile.
__global__ __launch_bounds__(128) void gemm_bf16_kernel(
    const u16* __restrict__ A, const u16* __restrict__ Wt,
    const float* __restrict__ res, float* __restrict__ outF, u16* __restrict__ outB,
    u16* __restrict__ outVT, int M, int N, int K, int actGelu)
{
  const int t = threadIdx.x, wave = t >> 5, lane = t & 31;
  const int m0 = blockIdx.y * 64 + (wave >> 1) * 32;
  const int n0 = blockIdx.x * 64 + (wave & 1) * 32;
  const int am  = lane & 15;
  const int akb = (lane < 16) ? 0 : 8;   // A fragment K base
  const int bn  = lane & 15;
  const int bkb = (lane < 16) ? 0 : 8;   // B fragment K base

  v8f acc[2][2] = {{{}, {}}, {{}, {}}};
  const u16* aRow[2] = { A + (size_t)(m0 + am) * K,       A + (size_t)(m0 + 16 + am) * K };
  const u16* bRow[2] = { Wt + (size_t)(n0 + bn) * K,      Wt + (size_t)(n0 + 16 + bn) * K };

  for (int k0 = 0; k0 < K; k0 += 32) {
    if (k0 + 32 < K) {  // prefetch next k-tile (global_prefetch_b8)
      __builtin_prefetch(aRow[0] + k0 + 32 + akb, 0, 3);
      __builtin_prefetch(aRow[1] + k0 + 32 + akb, 0, 3);
      __builtin_prefetch(bRow[0] + k0 + 32 + bkb, 0, 3);
      __builtin_prefetch(bRow[1] + k0 + 32 + bkb, 0, 3);
    }
    v16bf af[2], bf[2];
#pragma unroll
    for (int i2 = 0; i2 < 2; ++i2) {
      af[i2] = mk16(ld8(aRow[i2] + k0 + akb), ld8(aRow[i2] + k0 + akb + 16));
      bf[i2] = mk16(ld8(bRow[i2] + k0 + bkb), ld8(bRow[i2] + k0 + bkb + 16));
    }
#pragma unroll
    for (int i2 = 0; i2 < 2; ++i2)
#pragma unroll
      for (int j2 = 0; j2 < 2; ++j2)
        acc[i2][j2] = __builtin_amdgcn_wmma_f32_16x16x32_bf16(
            false, af[i2], false, bf[j2], (short)0, acc[i2][j2], false, false);
  }

  // epilogue; C/D layout: N=lane&15; rows r -> M = r + (lane<16 ? 0 : 8)
#pragma unroll
  for (int i2 = 0; i2 < 2; ++i2)
#pragma unroll
    for (int j2 = 0; j2 < 2; ++j2) {
      const int nB = n0 + j2 * 16 + (lane & 15);
      const int mB = m0 + i2 * 16 + ((lane < 16) ? 0 : 8);
      if (outVT) {
        // transposed store: 8 consecutive rows -> contiguous 16B along R
        v8bf pk;
#pragma unroll
        for (int r = 0; r < 8; ++r) pk[r] = (__bf16)acc[i2][j2][r];
        size_t vtb = (((size_t)(mB >> 10) * Hh + (nB >> 5)) * Dd + (nB & 31)) * Rr + (mB & (Rr - 1));
        *(uint4*)&outVT[vtb] = __builtin_bit_cast(uint4, pk);
      } else {
#pragma unroll
        for (int r = 0; r < 8; ++r) {
          size_t idx = (size_t)(mB + r) * N + nB;
          float v = acc[i2][j2][r];
          if (res) v += res[idx];
          if (actGelu) v = 0.5f * v * (1.0f + erff(v * 0.70710678118654752f));
          if (outF) outF[idx] = v;
          if (outB) st_bf16(&outB[idx], v);
        }
      }
    }
}

// ---------- feature-group attention: seq len G=8, per (b*r, head) block ----------
__global__ __launch_bounds__(64) void featattn_kernel(
    const u16* __restrict__ Q, const u16* __restrict__ K,
    const u16* __restrict__ V, u16* __restrict__ O)
{
  const int br = blockIdx.x, h = blockIdx.y;
  const int t = threadIdx.x;
  __shared__ float S[8][8];
  __shared__ float P[8][8];
  const int i = t >> 3, j = t & 7;
  const size_t base = (size_t)br * (Gg * Ee) + (size_t)h * Dd;  // + g*Ee + d

  const u16* qp = Q + base + (size_t)i * Ee;
  const u16* kp = K + base + (size_t)j * Ee;
  float acc = 0.f;
#pragma unroll
  for (int c = 0; c < 4; ++c) {
    v8bf qv = ld8(qp + c * 8), kv = ld8(kp + c * 8);
#pragma unroll
    for (int e = 0; e < 8; ++e) acc += (float)qv[e] * (float)kv[e];
  }
  S[i][j] = acc * 0.17677669529663687f;  // 1/sqrt(32)
  __syncthreads();

  if (t < 8) {
    float mx = -1e30f;
#pragma unroll
    for (int c = 0; c < 8; ++c) mx = fmaxf(mx, S[t][c]);
    float sum = 0.f;
#pragma unroll
    for (int c = 0; c < 8; ++c) { float p = expf(S[t][c] - mx); P[t][c] = p; sum += p; }
    float inv = 1.f / sum;
#pragma unroll
    for (int c = 0; c < 8; ++c) P[t][c] *= inv;
  }
  __syncthreads();

  // A·V: threads 0..31 each own (row i2, 8-wide d chunk) -> vector loads + one b128 store
  if (t < 32) {
    const int i2 = t >> 2, ch = (t & 3) * 8;
    float acc8[8] = {0.f, 0.f, 0.f, 0.f, 0.f, 0.f, 0.f, 0.f};
#pragma unroll
    for (int c = 0; c < 8; ++c) {
      v8bf vv = ld8(V + base + (size_t)c * Ee + ch);
      float pw = P[i2][c];
#pragma unroll
      for (int e = 0; e < 8; ++e) acc8[e] += pw * (float)vv[e];
    }
    v8bf pk;
#pragma unroll
    for (int e = 0; e < 8; ++e) pk[e] = (__bf16)acc8[e];
    *(uint4*)&O[base + (size_t)i2 * Ee + ch] = __builtin_bit_cast(uint4, pk);
  }
}

// ---------- row attention (flash-style WMMA), train/test split at sep ----------
// Q,K: [BG,R,E] bf16; Vt: [BG,H,D,R] bf16 (transposed). grid (R/16, H, BG), block 32.
__global__ __launch_bounds__(32) void rowattn_kernel(
    const u16* __restrict__ Q, const u16* __restrict__ Km,
    const u16* __restrict__ Vt, u16* __restrict__ O, int sep)
{
  const int r0 = blockIdx.x * 16;
  const int h  = blockIdx.y;
  const int bg = blockIdx.z;
  const int kvh = (r0 >= sep) ? 0 : h;
  const int lane = threadIdx.x;

  __shared__ __align__(16) float S[16][32];
  __shared__ float rowM[16], rowL[16], rowC[16];
  if (lane < 16) { rowM[lane] = -1e30f; rowL[lane] = 0.f; }

  const int am  = lane & 15;
  const int akb = (lane < 16) ? 0 : 8;
  const int bn  = lane & 15;
  const int bkb = (lane < 16) ? 0 : 8;
  const float scale = 0.17677669529663687f;

  // Q fragment: A 16x32 (M=query, K=channel d) — contiguous along d
  const u16* qp = Q + ((size_t)bg * Rr + (r0 + am)) * Ee + (size_t)h * Dd;
  v16bf aq = mk16(ld8(qp + akb), ld8(qp + akb + 16));

  // V^T row bases: lane's d-column is contiguous along keys
  const u16* vt0 = Vt + (((size_t)bg * Hh + kvh) * Dd + bn) * Rr;
  const u16* vt1 = vt0 + (size_t)16 * Rr;

  v8f o0 = {}, o1 = {};
  const v8f zeroC = {};

  for (int kc = 0; kc < sep; kc += 32) {
    // K^T fragments: lane = key, contiguous along d
    const u16* kp0 = Km + ((size_t)bg * Rr + (kc + bn)) * Ee + (size_t)kvh * Dd;
    const u16* kp1 = kp0 + (size_t)16 * Ee;
    v16bf bk0 = mk16(ld8(kp0 + bkb), ld8(kp0 + bkb + 16));
    v16bf bk1 = mk16(ld8(kp1 + bkb), ld8(kp1 + bkb + 16));

    v8f s0 = __builtin_amdgcn_wmma_f32_16x16x32_bf16(false, aq, false, bk0, (short)0, zeroC, false, false);
    v8f s1 = __builtin_amdgcn_wmma_f32_16x16x32_bf16(false, aq, false, bk1, (short)0, zeroC, false, false);

    // scatter S tile (16 queries x 32 keys) to LDS
    const int srow = (lane < 16) ? 0 : 8;
#pragma unroll
    for (int r = 0; r < 8; ++r) {
      S[srow + r][lane & 15]        = s0[r];
      S[srow + r][16 + (lane & 15)] = s1[r];
    }
    __syncthreads();

    // online softmax per row
    if (lane < 16) {
      float mOld = rowM[lane];
      float mx = mOld;
#pragma unroll
      for (int c = 0; c < 32; ++c) mx = fmaxf(mx, S[lane][c] * scale);
      float corr = expf(mOld - mx);
      float sum = 0.f;
#pragma unroll
      for (int c = 0; c < 32; ++c) {
        float p = expf(S[lane][c] * scale - mx);
        S[lane][c] = p;
        sum += p;
      }
      rowL[lane] = rowL[lane] * corr + sum;
      rowM[lane] = mx;
      rowC[lane] = corr;
    }
    __syncthreads();

    // rescale accumulators by per-row correction
    const int mrb = (lane < 16) ? 0 : 8;
#pragma unroll
    for (int r = 0; r < 8; ++r) {
      float c = rowC[mrb + r];
      o0[r] *= c;
      o1[r] *= c;
    }

    // P fragment from LDS (contiguous runs along keys)
    v16bf ap;
#pragma unroll
    for (int half = 0; half < 2; ++half) {
      const float* src = &S[am][akb + half * 16];
#pragma unroll
      for (int e = 0; e < 8; ++e)
        ap[half * 8 + e] = (__bf16)src[e];
    }

    // V fragments from V^T: contiguous along keys
    v16bf bv0 = mk16(ld8(vt0 + kc + bkb), ld8(vt0 + kc + bkb + 16));
    v16bf bv1 = mk16(ld8(vt1 + kc + bkb), ld8(vt1 + kc + bkb + 16));

    o0 = __builtin_amdgcn_wmma_f32_16x16x32_bf16(false, ap, false, bv0, (short)0, o0, false, false);
    o1 = __builtin_amdgcn_wmma_f32_16x16x32_bf16(false, ap, false, bv1, (short)0, o1, false, false);
    __syncthreads();
  }

  // normalize and store
  const int mrb = (lane < 16) ? 0 : 8;
#pragma unroll
  for (int r = 0; r < 8; ++r) {
    float inv = 1.f / rowL[mrb + r];
    size_t ob = ((size_t)bg * Rr + (r0 + mrb + r)) * Ee + (size_t)h * Dd;
    st_bf16(&O[ob + (lane & 15)],      o0[r] * inv);
    st_bf16(&O[ob + 16 + (lane & 15)], o1[r] * inv);
  }
}

// ---------- RMSNorm with optional layout transpose folded into writeback ----------
// mode 0: [X,E]->[X,E]; mode 1: [B,R,G,E]->[B,G,R,E]; mode 2: [B,G,R,E]->[B,R,G,E]
__global__ __launch_bounds__(64) void rmsnorm_kernel(
    const float* __restrict__ in, const float* __restrict__ gw,
    float* __restrict__ outF, u16* __restrict__ outB, int mode)
{
  __shared__ float red[64];
  const int tk = blockIdx.x;
  const int lane = threadIdx.x;
  const size_t inBase = (size_t)tk * Ee;
  size_t outBase;
  if (mode == 0) {
    outBase = inBase;
  } else if (mode == 1) {
    int b = tk >> 13, r = (tk >> 3) & (Rr - 1), g = tk & (Gg - 1);
    outBase = (((size_t)b * Gg + g) * Rr + r) * Ee;
  } else {
    int b = tk >> 13, g = (tk >> 10) & (Gg - 1), r = tk & (Rr - 1);
    outBase = (((size_t)b * Rr + r) * Gg + g) * Ee;
  }

  float v0 = in[inBase + lane];
  float v1 = in[inBase + lane + 64];
  float v2 = in[inBase + lane + 128];
  red[lane] = v0 * v0 + v1 * v1 + v2 * v2;
  __syncthreads();
  for (int s = 32; s > 0; s >>= 1) {
    if (lane < s) red[lane] += red[lane + s];
    __syncthreads();
  }
  const float scale = rsqrtf(red[0] * (1.0f / (float)Ee) + 1.1920929e-07f);

  float vv[3] = {v0, v1, v2};
#pragma unroll
  for (int q = 0; q < 3; ++q) {
    int e = lane + q * 64;
    float o = vv[q] * scale * gw[e];
    if (outF) outF[outBase + e] = o;
    if (outB) st_bf16(&outB[outBase + e], o);
  }
}

// ============================ host launcher ============================
extern "C" void kernel_launch(void* const* d_in, const int* in_sizes, int n_in,
                              void* d_out, int out_size, void* d_ws, size_t ws_size,
                              hipStream_t stream) {
  (void)in_sizes; (void)n_in; (void)out_size; (void)ws_size;

  const float* hidden = (const float*)d_in[0];
  const float* Wsrc[10];
  for (int i = 0; i < 10; ++i) Wsrc[i] = (const float*)d_in[1 + i];
  const float* g1 = (const float*)d_in[11];
  const float* g2 = (const float*)d_in[12];
  const float* g3 = (const float*)d_in[13];

  // carve workspace
  char* p = (char*)d_ws;
  auto carve = [&](size_t bytes) {
    char* r = p;
    p += (bytes + 255) & ~(size_t)255;
    return r;
  };
  const size_t ACT_F32 = (size_t)NT * Ee * sizeof(float);
  const size_t ACT_B16 = (size_t)NT * Ee * sizeof(u16);

  float* X   = (float*)carve(ACT_F32);
  float* XF  = (float*)carve(ACT_F32);   // reused: xf, xc_pre, mlp-out
  float* XR  = (float*)carve(ACT_F32);
  float* XC  = (float*)carve(ACT_F32);
  u16*   XB  = (u16*)carve(ACT_B16);
  u16*   QB  = (u16*)carve(ACT_B16);
  u16*   KB  = (u16*)carve(ACT_B16);     // contiguous with QB
  u16*   VB  = (u16*)carve(ACT_B16);     // stage1: [BR,G,E]; stage2: V^T [BG,H,D,R]
  u16*   AB  = (u16*)carve(ACT_B16);
  u16*   XRB = (u16*)carve(ACT_B16);
  u16*   XCB = (u16*)carve(ACT_B16);
  u16*   H1B = QB;                       // overlay QB+KB (dead during MLP), NT*Ff*2 bytes

  const size_t wEE = (size_t)Ll * Ee * Ee;
  const size_t wEF = (size_t)Ll * Ee * Ff;
  u16* Wb[10];
  for (int i = 0; i < 10; ++i) Wb[i] = (u16*)carve(((i < 8) ? wEE : wEF) * sizeof(u16));

  // convert + transpose weights once per call: Wt[l][N][K]
  for (int i = 0; i < 10; ++i) {
    int Kd = (i == 9) ? Ff : Ee;               // W2 is [F,E]
    int Nd = (i == 8) ? Ff : Ee;               // W1 is [E,F]
    int n = (int)((i < 8) ? wEE : wEF);
    cvt_tr_kernel<<<(n + 255) / 256, 256, 0, stream>>>(Wsrc[i], Wb[i], Kd, Nd, n);
  }
  {
    int n = NT * Ee;
    copy_cvt_kernel<<<(n + 255) / 256, 256, 0, stream>>>(hidden, X, XB, n);
  }

  const dim3 gN192(Ee / 64, NT / 64);     // N=192 GEMM grid (3 x 512)
  const dim3 gN384(Ff / 64, NT / 64);     // N=384 GEMM grid (6 x 512)

  for (int i = 0; i < Ll; ++i) {
    const u16* wq1 = Wb[0] + (size_t)i * Ee * Ee;
    const u16* wk1 = Wb[1] + (size_t)i * Ee * Ee;
    const u16* wv1 = Wb[2] + (size_t)i * Ee * Ee;
    const u16* wo1 = Wb[3] + (size_t)i * Ee * Ee;
    const u16* wq2 = Wb[4] + (size_t)i * Ee * Ee;
    const u16* wk2 = Wb[5] + (size_t)i * Ee * Ee;
    const u16* wv2 = Wb[6] + (size_t)i * Ee * Ee;
    const u16* wo2 = Wb[7] + (size_t)i * Ee * Ee;
    const u16* w1  = Wb[8] + (size_t)i * Ee * Ff;
    const u16* w2  = Wb[9] + (size_t)i * Ff * Ee;

    // --- feature-group attention ---
    gemm_bf16_kernel<<<gN192, 128, 0, stream>>>(XB, wq1, nullptr, nullptr, QB, nullptr, NT, Ee, Ee, 0);
    gemm_bf16_kernel<<<gN192, 128, 0, stream>>>(XB, wk1, nullptr, nullptr, KB, nullptr, NT, Ee, Ee, 0);
    gemm_bf16_kernel<<<gN192, 128, 0, stream>>>(XB, wv1, nullptr, nullptr, VB, nullptr, NT, Ee, Ee, 0);
    featattn_kernel<<<dim3(BRn, Hh), 64, 0, stream>>>(QB, KB, VB, AB);
    gemm_bf16_kernel<<<gN192, 128, 0, stream>>>(AB, wo1, X, XF, nullptr, nullptr, NT, Ee, Ee, 0);
    rmsnorm_kernel<<<NT, 64, 0, stream>>>(XF, g1 + (size_t)i * Ee, XR, XRB, 1);

    // --- row attention (train/test split) ---
    gemm_bf16_kernel<<<gN192, 128, 0, stream>>>(XRB, wq2, nullptr, nullptr, QB, nullptr, NT, Ee, Ee, 0);
    gemm_bf16_kernel<<<gN192, 128, 0, stream>>>(XRB, wk2, nullptr, nullptr, KB, nullptr, NT, Ee, Ee, 0);
    gemm_bf16_kernel<<<gN192, 128, 0, stream>>>(XRB, wv2, nullptr, nullptr, nullptr, VB, NT, Ee, Ee, 0);
    rowattn_kernel<<<dim3(Rr / 16, Hh, BGn), 32, 0, stream>>>(QB, KB, VB, AB, SEP);
    gemm_bf16_kernel<<<gN192, 128, 0, stream>>>(AB, wo2, XR, XF, nullptr, nullptr, NT, Ee, Ee, 0);
    rmsnorm_kernel<<<NT, 64, 0, stream>>>(XF, g2 + (size_t)i * Ee, XC, XCB, 0);

    // --- MLP (exact GELU) + final RMSNorm (with transpose back) ---
    gemm_bf16_kernel<<<gN384, 128, 0, stream>>>(XCB, w1, nullptr, nullptr, H1B, nullptr, NT, Ff, Ee, 1);
    gemm_bf16_kernel<<<gN192, 128, 0, stream>>>(H1B, w2, XC, XF, nullptr, nullptr, NT, Ee, Ff, 0);
    float* outX = (i == Ll - 1) ? (float*)d_out : X;
    rmsnorm_kernel<<<NT, 64, 0, stream>>>(XF, g3 + (size_t)i * Ee, outX, XB, 2);
  }
}
